// Gate_8650064134723
// MI455X (gfx1250) — matrix operand order repaired and checked by reading, not compiled, not run on hardware
//
#include <hip/hip_runtime.h>
#include <hip/hip_bf16.h>

// ---------------------------------------------------------------------------
// Fused MoE-gate: x[8192,4096] -> fc1(64) -> BN(eval)+ReLU -> fc2(64)
//                 -> top2 -> softmax -> scatter gates + indices.
//
// Memory-bound: 134 MB of x @ 23.3 TB/s ~= 5.8 us floor. The x / w1 tile
// staging runs on the CDNA5 Tensor Data Mover (TENSOR_LOAD_TO_LDS,
// TENSORcnt-tracked, double-buffered) so the global->LDS stream is fully
// asynchronous to the bf16 WMMA pipeline -- no VGPR round trip, no
// one-load-at-a-time s_wait_loadcnt serialization. TDM's D# pad_interval /
// pad_amount adds per-row LDS padding for bank-conflict-free ds_load_b128
// fragment reads.
//
// Requires ws_size >= 64*4096*2 bytes (512 KB) for the bf16 copy of w1.
// ---------------------------------------------------------------------------

namespace {

constexpr int kN = 8192;
constexpr int kD = 4096;
constexpr int kH = 64;
constexpr int kE = 64;
constexpr float kEps = 1e-5f;

constexpr int BM      = 64;            // rows per workgroup
constexpr int KB      = 64;            // K elements staged per chunk
constexpr int WAVES   = 4;             // wave32 waves per workgroup
constexpr int THREADS = WAVES * 32;    // 128 threads
constexpr int kChunks = kD / KB;       // 64

// LDS strides produced by TDM padding (elements)
constexpr int XTF_STRIDE = 68;  // f32 row stride: 64 + 4 pad (272 B, 16B aligned)
constexpr int WT_STRIDE  = 72;  // bf16 row stride: 64 + 8 pad (144 B, 16B aligned)
constexpr int HT_STRIDE  = 72;  // bf16 hidden tile row stride
constexpr int LG_STRIDE  = 68;  // f32 logits row stride

// LDS region offsets (bytes)
constexpr int XA_OFF  = 0;                       // 64*272 = 17408
constexpr int XB_OFF  = 17408;                   // ends 34816
constexpr int W1A_OFF = 34816;                   // 64*144 = 9216
constexpr int W1B_OFF = 44032;                   // ends 53248
constexpr int W2_OFF  = 53248;                   // 9216, ends 62464
constexpr int SMEM_BYTES = 62464;
// hT (bf16, 9216 B) and logits (17408 B) reuse region [0, 17408) after the
// main loop, guarded by barriers.

typedef __attribute__((ext_vector_type(16))) __bf16 v16bf;
typedef __attribute__((ext_vector_type(8)))  __bf16 v8bf;
typedef __attribute__((ext_vector_type(4)))  __bf16 v4bf;
typedef __attribute__((ext_vector_type(8)))  float  v8f;
typedef __attribute__((ext_vector_type(4)))  unsigned int v4u;
typedef __attribute__((ext_vector_type(8)))  int v8i;
typedef __attribute__((ext_vector_type(4)))  int v4i;

// ---------------------------------------------------------------------------
// Tensor Data Mover: 2D tile load, global -> LDS, with LDS row padding.
// D# layout per CDNA5 ISA. This toolchain exposes the
// 6-arg builtin: (uint32x4 g0, int32x8 g1, int32x4, int32x4, int32x8, i32 cpol).
//   data_size_code: 0=1B 1=2B 2=4B 3=8B
//   pad_interval_code: pad after 2^(code+1) DWORDs; pad_amount_code: (code+1) DWORDs
// ---------------------------------------------------------------------------
__device__ __forceinline__ void tdm_load_2d(unsigned int lds_off,
                                            const void* gptr,
                                            unsigned int tensor_dim0,
                                            unsigned int tensor_dim1,
                                            unsigned long long dim0_stride,
                                            unsigned int tile_dim0,
                                            unsigned int tile_dim1,
                                            unsigned int data_size_code,
                                            unsigned int pad_interval_code,
                                            unsigned int pad_amount_code) {
  const unsigned long long ga = (unsigned long long)(uintptr_t)gptr;
  v4u g0;
  g0[0] = 1u;                                   // count=1 (valid), user mode
  g0[1] = lds_off;                              // LDS byte address (addr[31:0])
  g0[2] = (unsigned int)ga;                     // global_addr[31:0]
  g0[3] = (unsigned int)((ga >> 32) & 0x1FFFFFFu) | (2u << 30);  // [56:32] | type=2
  v8i g1;
  g1[0] = (int)((data_size_code << 16) | (1u << 20) |            // pad_enable
                (pad_interval_code << 22) | (pad_amount_code << 25));
  g1[1] = (int)((tensor_dim0 & 0xFFFFu) << 16);                  // abar=0 | dim0 lo16
  g1[2] = (int)(((tensor_dim0 >> 16) & 0xFFFFu) | ((tensor_dim1 & 0xFFFFu) << 16));
  g1[3] = (int)(((tensor_dim1 >> 16) & 0xFFFFu) | ((tile_dim0 & 0xFFFFu) << 16));
  g1[4] = (int)(tile_dim1 & 0xFFFFu);                            // tile_dim2 = 0
  g1[5] = (int)(unsigned int)(dim0_stride & 0xFFFFFFFFull);
  g1[6] = (int)(unsigned int)((dim0_stride >> 32) & 0xFFFFull);  // dim1_stride = 0 (2D)
  g1[7] = 0;
  const v4i z4 = {0, 0, 0, 0};
  const v8i z8 = {0, 0, 0, 0, 0, 0, 0, 0};
  __builtin_amdgcn_tensor_load_to_lds(g0, g1, z4, z4, z8, 0);
}

__device__ __forceinline__ void tdm_load_x_tile(unsigned int lds_off, const float* g) {
  // 64x64 f32 tile of x[8192,4096]; pad: 64 DW interval (code 5), 4 DW (code 3)
  tdm_load_2d(lds_off, g, kD, kN, (unsigned long long)kD, KB, BM, 2u, 5u, 3u);
}

__device__ __forceinline__ void tdm_load_w1_tile(unsigned int lds_off, const __bf16* g) {
  // 64x64 bf16 tile of w1bf[64,4096]; pad: 32 DW interval (code 4), 4 DW (code 3)
  tdm_load_2d(lds_off, g, kD, kH, (unsigned long long)kD, KB, kH, 1u, 4u, 3u);
}

// A-fragment (16x32, ISA 7.12.2) built from the fp32 x tile with on-the-fly
// bf16 conversion: lanes 0-15 take K[0:8)+K[16:24), lanes 16-31 K[8:16)+K[24:32).
__device__ __forceinline__ v16bf load_frag_a_f32(const float* rowPtr, int lhalf) {
  const float4 f0 = *(const float4*)(rowPtr + lhalf * 8);
  const float4 f1 = *(const float4*)(rowPtr + lhalf * 8 + 4);
  const float4 f2 = *(const float4*)(rowPtr + 16 + lhalf * 8);
  const float4 f3 = *(const float4*)(rowPtr + 16 + lhalf * 8 + 4);
  v16bf a = {(__bf16)f0.x, (__bf16)f0.y, (__bf16)f0.z, (__bf16)f0.w,
             (__bf16)f1.x, (__bf16)f1.y, (__bf16)f1.z, (__bf16)f1.w,
             (__bf16)f2.x, (__bf16)f2.y, (__bf16)f2.z, (__bf16)f2.w,
             (__bf16)f3.x, (__bf16)f3.y, (__bf16)f3.z, (__bf16)f3.w};
  return a;
}

// A-fragment from a bf16 tile (used for fc2's hidden tile).
__device__ __forceinline__ v16bf load_frag_a_bf16(const __bf16* rowPtr, int lhalf) {
  const v8bf lo = *(const v8bf*)(rowPtr + lhalf * 8);
  const v8bf hi = *(const v8bf*)(rowPtr + 16 + lhalf * 8);
  return __builtin_shufflevector(lo, hi, 0, 1, 2, 3, 4, 5, 6, 7,
                                 8, 9, 10, 11, 12, 13, 14, 15);
}

// B-fragment (32x16): lanes 0-15 hold K[0:8)+K[16:24), lanes 16-31 K[8:16)+K[24:32) for
// column N = lane%16 (weights stored [n][k] row-major in LDS).
__device__ __forceinline__ v16bf load_frag_b(const __bf16* rowPtr, int lhalf) {
  const v8bf lo = *(const v8bf*)(rowPtr + lhalf * 8);
  const v8bf hi = *(const v8bf*)(rowPtr + 16 + lhalf * 8);
  return __builtin_shufflevector(lo, hi, 0, 1, 2, 3, 4, 5, 6, 7,
                                 8, 9, 10, 11, 12, 13, 14, 15);
}

// ---------------------------------------------------------------------------
// Prep kernel: w1 f32 -> bf16 (1 MB -> 512 KB in d_ws), run once per launch.
// ---------------------------------------------------------------------------
__global__ __launch_bounds__(256) void convert_w1_bf16(const float* __restrict__ w1,
                                                       __bf16* __restrict__ out) {
  const int i = (blockIdx.x * 256 + threadIdx.x) * 4;
  const float4 v = *(const float4*)(w1 + i);
  v4bf b = {(__bf16)v.x, (__bf16)v.y, (__bf16)v.z, (__bf16)v.w};
  *(v4bf*)(out + i) = b;
}

// ---------------------------------------------------------------------------
// Main fused kernel.
// ---------------------------------------------------------------------------
__global__ __launch_bounds__(THREADS) void moe_gate_fused_kernel(
    const float* __restrict__ x, const __bf16* __restrict__ w1bf,
    const float* __restrict__ b1, const float* __restrict__ gamma,
    const float* __restrict__ beta, const float* __restrict__ rmean,
    const float* __restrict__ rvar, const float* __restrict__ w2,
    const float* __restrict__ b2, float* __restrict__ gatesOut,
    int* __restrict__ idxOut) {
  __shared__ __align__(16) unsigned char smem[SMEM_BYTES];
  const unsigned int smemBase = (unsigned int)(uintptr_t)(void*)smem;  // LDS addr[31:0]

  const int tid   = threadIdx.x;
  const int lane  = tid & 31;
  const int wave  = tid >> 5;
  const int lhalf = lane >> 4;
  const int lmod  = lane & 15;
  const int rowBase = blockIdx.x * BM;

  // Phase 0: stage w2 [E][H] as bf16 (batched loads first for MLP, then cvt+store)
  {
    __bf16* w2T = (__bf16*)(smem + W2_OFF);
    float4 v[8];
#pragma unroll
    for (int i = 0; i < 8; ++i) {
      const int fid = tid + i * THREADS;
      v[i] = *(const float4*)(w2 + (fid >> 4) * kH + (fid & 15) * 4);
    }
#pragma unroll
    for (int i = 0; i < 8; ++i) {
      const int fid = tid + i * THREADS;
      v4bf b = {(__bf16)v[i].x, (__bf16)v[i].y, (__bf16)v[i].z, (__bf16)v[i].w};
      *(v4bf*)(w2T + (fid >> 4) * WT_STRIDE + (fid & 15) * 4) = b;
    }
  }

  // Phase 1: h = x @ w1^T. TDM double-buffers the tiles; waves run bf16 WMMA.
  v8f acc[4] = {};
  const float* xBlock = x + (size_t)rowBase * kD;

  if (wave == 0) {  // TDM ops are per-wave (EXEC ignored); one wave issues them
    tdm_load_x_tile(smemBase + XA_OFF, xBlock);
    tdm_load_w1_tile(smemBase + W1A_OFF, w1bf);
  }

  for (int c = 0; c < kChunks; ++c) {
    const int cur = c & 1;
    if (wave == 0) {
      if (c + 1 < kChunks) {
        const int nxt = (c + 1) & 1;
        // Overwrites the buffer last read in iter c-1; all waves passed that
        // iteration's trailing barrier, so WAR-safe.
        tdm_load_x_tile(smemBase + (nxt ? XB_OFF : XA_OFF),
                        xBlock + (size_t)(c + 1) * KB);
        tdm_load_w1_tile(smemBase + (nxt ? W1B_OFF : W1A_OFF),
                         w1bf + (size_t)(c + 1) * KB);
        // TENSORcnt is in-order: <=2 outstanding => chunk c's pair completed.
        __builtin_amdgcn_s_wait_tensorcnt(2);
      } else {
        __builtin_amdgcn_s_wait_tensorcnt(0);
      }
    }
    __syncthreads();  // publish chunk c's LDS data to all waves

    const float* xTf = (const float*)(smem + (cur ? XB_OFF : XA_OFF));
    const __bf16* w1T = (const __bf16*)(smem + (cur ? W1B_OFF : W1A_OFF));
    const float* aRow = xTf + (wave * 16 + lmod) * XTF_STRIDE;
#pragma unroll
    for (int ks = 0; ks < KB; ks += 32) {
      const v16bf a = load_frag_a_f32(aRow + ks, lhalf);
#pragma unroll
      for (int nt = 0; nt < 4; ++nt) {
        const v16bf b = load_frag_b(w1T + (nt * 16 + lmod) * WT_STRIDE + ks, lhalf);
        acc[nt] = __builtin_amdgcn_wmma_f32_16x16x32_bf16(
            false, a, false, b, (short)0, acc[nt], false, false);
      }
    }
    __syncthreads();  // all waves done reading buf[cur]; next iter may overwrite other buf
  }

  // Phase 2: bias + BN(eval) + ReLU -> bf16 hidden tile (reuses x region)
  __bf16* hT = (__bf16*)smem;
#pragma unroll
  for (int nt = 0; nt < 4; ++nt) {
    const int col = nt * 16 + lmod;
    const float sc = gamma[col] * rsqrtf(rvar[col] + kEps);
    const float sh = beta[col] - rmean[col] * sc;
    const float bb = b1[col];
#pragma unroll
    for (int j = 0; j < 8; ++j) {
      const int m = j + lhalf * 8;  // C-layout: lanes<16 -> M=j, lanes>=16 -> M=j+8
      float v = (acc[nt][j] + bb) * sc + sh;
      v = fmaxf(v, 0.0f);
      hT[(wave * 16 + m) * HT_STRIDE + col] = (__bf16)v;
    }
  }

  // Phase 3: logits = h @ w2^T (K = 64 -> two WMMA steps), + b2
  v8f lacc[4] = {};
  const __bf16* w2T = (const __bf16*)(smem + W2_OFF);
  const __bf16* aRow2 = hT + (wave * 16 + lmod) * HT_STRIDE;
#pragma unroll
  for (int ks = 0; ks < kH; ks += 32) {
    const v16bf a = load_frag_a_bf16(aRow2 + ks, lhalf);
#pragma unroll
    for (int nt = 0; nt < 4; ++nt) {
      const v16bf b = load_frag_b(w2T + (nt * 16 + lmod) * WT_STRIDE + ks, lhalf);
      lacc[nt] = __builtin_amdgcn_wmma_f32_16x16x32_bf16(
          false, a, false, b, (short)0, lacc[nt], false, false);
    }
  }
  __syncthreads();  // all waves done reading hT; logits region overlaps it

  float* lgT = (float*)smem;
#pragma unroll
  for (int nt = 0; nt < 4; ++nt) {
    const int col = nt * 16 + lmod;
    const float bias2 = b2[col];
#pragma unroll
    for (int j = 0; j < 8; ++j) {
      const int r = wave * 16 + j + lhalf * 8;
      lgT[r * LG_STRIDE + col] = lacc[nt][j] + bias2;
    }
  }
  __syncthreads();

  // Phase 4: per-row top-2 + 2-way softmax + scatter (first BM threads)
  if (tid < BM) {
    const float* row = lgT + tid * LG_STRIDE;
    float v0 = -INFINITY, v1 = -INFINITY;
    int i0 = 0, i1 = 0;
#pragma unroll 4
    for (int e = 0; e < kE; ++e) {
      const float v = row[e];
      if (v > v0) { v1 = v0; i1 = i0; v0 = v; i0 = e; }
      else if (v > v1) { v1 = v; i1 = e; }
    }
    const float g0 = 1.0f / (1.0f + __expf(v1 - v0));  // softmax of [v0, v1]
    const float g1 = 1.0f - g0;
    const size_t gr = (size_t)rowBase + tid;
    float* gout = gatesOut + gr * kE;
#pragma unroll 4
    for (int e = 0; e < kE; ++e) {
      const float g = (e == i0) ? g0 : ((e == i1) ? g1 : 0.0f);
      __builtin_nontemporal_store(g, gout + e);  // streamed once, keep out of caches
    }
    int* iout = idxOut + gr * 2;
    __builtin_nontemporal_store(i0, iout + 0);
    __builtin_nontemporal_store(i1, iout + 1);
  }
}

}  // namespace

extern "C" void kernel_launch(void* const* d_in, const int* in_sizes, int n_in,
                              void* d_out, int out_size, void* d_ws, size_t ws_size,
                              hipStream_t stream) {
  (void)in_sizes; (void)n_in; (void)out_size; (void)ws_size;
  const float* x     = (const float*)d_in[0];
  const float* w1    = (const float*)d_in[1];
  const float* b1    = (const float*)d_in[2];
  const float* gamma = (const float*)d_in[3];
  const float* beta  = (const float*)d_in[4];
  const float* rmean = (const float*)d_in[5];
  const float* rvar  = (const float*)d_in[6];
  const float* w2    = (const float*)d_in[7];
  const float* b2    = (const float*)d_in[8];

  float* gates = (float*)d_out;                    // [N, E] fp32
  int* topIdx  = (int*)(gates + (size_t)kN * kE);  // [N, 2] int32, concatenated
  __bf16* w1bf = (__bf16*)d_ws;                    // 512 KB scratch

  // Prep: w1 -> bf16 (64*4096 elems, 4 per thread)
  hipLaunchKernelGGL(convert_w1_bf16, dim3((kH * kD) / (256 * 4)), dim3(256), 0,
                     stream, w1, w1bf);

  // Main fused kernel: 128 workgroups x 4 wave32 waves
  hipLaunchKernelGGL(moe_gate_fused_kernel, dim3(kN / BM), dim3(THREADS), 0, stream,
                     x, w1bf, b1, gamma, beta, rmean, rvar, w2, b2, gates, topIdx);
}
